// GatedLinearStateAttention_15564961481276
// MI455X (gfx1250) — matrix operand order repaired and checked by reading, not compiled, not run on hardware
//
#include <hip/hip_runtime.h>

#define NT   4096   // B*T token rows
#define DIMC 1024
#define NH   16
#define HD   64
#define TSEQ 1024
#define BB   4

typedef __attribute__((ext_vector_type(16))) _Float16 v16h;
typedef __attribute__((ext_vector_type(8)))  _Float16 v8h;
typedef __attribute__((ext_vector_type(8)))  float    v8f;
typedef __attribute__((ext_vector_type(4)))  int      v4i;

// gfx1250 async global->LDS copy path (ASYNCcnt-tracked), when the toolchain
// declares the builtins. Host pass and older toolchains take the fallback.
#if __has_builtin(__builtin_amdgcn_global_load_async_to_lds_b128) && \
    __has_builtin(__builtin_amdgcn_s_wait_asynccnt)
#define USE_ASYNC_LDS 1
#else
#define USE_ASYNC_LDS 0
#endif

#if USE_ASYNC_LDS
typedef __attribute__((address_space(1))) v4i as1_v4i;   // __device__ int4
typedef __attribute__((address_space(3))) v4i as3_v4i;   // __shared__ int4
// one 16-byte lane-level async transfer: global -> LDS
static __device__ __forceinline__ void async_copy16(const _Float16* g, _Float16* l) {
  __builtin_amdgcn_global_load_async_to_lds_b128(
      (as1_v4i*)(unsigned long long)(const void*)g,
      (as3_v4i*)(unsigned)(unsigned long long)(void*)l, 0, 0);
}
static __device__ __forceinline__ void async_wait0() {
  __builtin_amdgcn_s_wait_asynccnt(0);
}
#endif

static __device__ __forceinline__ v16h pack16(v8h lo, v8h hi) {
  v16h r;
#pragma unroll
  for (int i = 0; i < 8; ++i) { r[i] = lo[i]; r[i + 8] = hi[i]; }
  return r;
}

static __device__ __forceinline__ v8f wmma_f16(v16h a, v16h b, v8f c) {
  // D = A(16x32 f16) x B(32x16 f16) + C(f32), wave32 CDNA5 WMMA
  return __builtin_amdgcn_wmma_f32_16x16x32_f16(false, a, false, b, (short)0, c,
                                                false, false);
}

// ---------------------------------------------------------------- converts
__global__ void f32_to_f16_kernel(const float* __restrict__ X,
                                  _Float16* __restrict__ Y, int n) {
  int i = blockIdx.x * blockDim.x + threadIdx.x;
  int stride = gridDim.x * blockDim.x;
  for (; i < n; i += stride) Y[i] = (_Float16)X[i];
}

// ---------------------------------------------------------------- GEMM
// Y[m,n] = sum_k A[m,k]*W[n,k] + bias[n]; A:[M,K] f16, W:[N,K] f16, Y:[M,N] f32
// Block 256 thr = 8 waves; tile 128(M) x 128(N) x 32(K); wave does 32x64.
// Double-buffered LDS; next K-slab streams (async-to-LDS when available)
// while WMMAs consume the current one.
__global__ __launch_bounds__(256) void gemm_xwT(const _Float16* __restrict__ A,
                                                const _Float16* __restrict__ W,
                                                const float* __restrict__ bias,
                                                float* __restrict__ Y,
                                                int M, int N, int K) {
  (void)M;
  __shared__ alignas(16) _Float16 lA[2][128][32];
  __shared__ alignas(16) _Float16 lB[2][128][32];

  const int tid  = threadIdx.x;
  const int lane = tid & 31;
  const int wave = tid >> 5;
  const int Mblk = blockIdx.y * 128;
  const int Nblk = blockIdx.x * 128;
  const int mw = (wave >> 1) * 32;   // 0,32,64,96
  const int nw = (wave & 1) * 64;    // 0,64

  const v8f zero = {0.f, 0.f, 0.f, 0.f, 0.f, 0.f, 0.f, 0.f};
  v8f acc[2][4];
#pragma unroll
  for (int i = 0; i < 2; ++i)
#pragma unroll
    for (int j = 0; j < 4; ++j) acc[i][j] = zero;

  const int lrow = tid >> 1;         // 0..127
  const int lk   = (tid & 1) * 16;   // 0 / 16

  const int am = lane & 15;
  const int ak = (lane < 16) ? 0 : 8;
  const int bn = lane & 15;
  const int bk = (lane < 16) ? 0 : 8;

  const _Float16* gA0 = A + (size_t)(Mblk + lrow) * K + lk;
  const _Float16* gB0 = W + (size_t)(Nblk + lrow) * K + lk;
  const int nIter = K / 32;

#if USE_ASYNC_LDS
  // prologue: slab 0 -> buffer 0 (ASYNCcnt-tracked lane copies)
  async_copy16(gA0, &lA[0][lrow][lk]);
  async_copy16(gA0 + 8, &lA[0][lrow][lk + 8]);
  async_copy16(gB0, &lB[0][lrow][lk]);
  async_copy16(gB0 + 8, &lB[0][lrow][lk + 8]);
#else
  v8h ra0 = *(const v8h*)(gA0);
  v8h ra1 = *(const v8h*)(gA0 + 8);
  v8h rb0 = *(const v8h*)(gB0);
  v8h rb1 = *(const v8h*)(gB0 + 8);
  *(v8h*)(&lA[0][lrow][lk])     = ra0;
  *(v8h*)(&lA[0][lrow][lk + 8]) = ra1;
  *(v8h*)(&lB[0][lrow][lk])     = rb0;
  *(v8h*)(&lB[0][lrow][lk + 8]) = rb1;
#endif

  for (int it = 0; it < nIter; ++it) {
    const int buf = it & 1;
#if USE_ASYNC_LDS
    async_wait0();               // this wave's fills of lA/lB[buf] complete
#endif
    __syncthreads();             // all waves' fills visible; prev reads done
    const bool more = (it + 1 < nIter);
#if USE_ASYNC_LDS
    if (more) {                  // stream next slab while we compute
      const _Float16* na = gA0 + (size_t)(it + 1) * 32;
      const _Float16* nb = gB0 + (size_t)(it + 1) * 32;
      async_copy16(na, &lA[buf ^ 1][lrow][lk]);
      async_copy16(na + 8, &lA[buf ^ 1][lrow][lk + 8]);
      async_copy16(nb, &lB[buf ^ 1][lrow][lk]);
      async_copy16(nb + 8, &lB[buf ^ 1][lrow][lk + 8]);
    }
#else
    if (more) {                  // loads in flight across the WMMA block
      const _Float16* na = gA0 + (size_t)(it + 1) * 32;
      const _Float16* nb = gB0 + (size_t)(it + 1) * 32;
      __builtin_prefetch(na + 32, 0, 1);
      __builtin_prefetch(nb + 32, 0, 1);
      ra0 = *(const v8h*)(na);
      ra1 = *(const v8h*)(na + 8);
      rb0 = *(const v8h*)(nb);
      rb1 = *(const v8h*)(nb + 8);
    }
#endif

    v16h afrag[2];
#pragma unroll
    for (int i = 0; i < 2; ++i) {
      v8h lo = *(const v8h*)(&lA[buf][mw + i * 16 + am][ak]);
      v8h hi = *(const v8h*)(&lA[buf][mw + i * 16 + am][ak + 16]);
      afrag[i] = pack16(lo, hi);
    }
    v16h bfrag[4];
#pragma unroll
    for (int j = 0; j < 4; ++j) {
      v8h lo = *(const v8h*)(&lB[buf][nw + j * 16 + bn][bk]);
      v8h hi = *(const v8h*)(&lB[buf][nw + j * 16 + bn][bk + 16]);
      bfrag[j] = pack16(lo, hi);
    }
#pragma unroll
    for (int i = 0; i < 2; ++i)
#pragma unroll
      for (int j = 0; j < 4; ++j)
        acc[i][j] = wmma_f16(afrag[i], bfrag[j], acc[i][j]);

#if !USE_ASYNC_LDS
    if (more) {
      *(v8h*)(&lA[buf ^ 1][lrow][lk])     = ra0;
      *(v8h*)(&lA[buf ^ 1][lrow][lk + 8]) = ra1;
      *(v8h*)(&lB[buf ^ 1][lrow][lk])     = rb0;
      *(v8h*)(&lB[buf ^ 1][lrow][lk + 8]) = rb1;
    }
#endif
  }

  const int cn = lane & 15;
  const int chalf = lane >> 4;
#pragma unroll
  for (int i = 0; i < 2; ++i)
#pragma unroll
    for (int j = 0; j < 4; ++j)
#pragma unroll
      for (int r = 0; r < 8; ++r) {
        int row = Mblk + mw + i * 16 + r + chalf * 8;
        int col = Nblk + nw + j * 16 + cn;
        Y[(size_t)row * N + col] = acc[i][j][r] + bias[col];
      }
}

// ---------------------------------------------------------------- gate
// g[row,h] = exp(-softplus(x[row,:] . Wg[h,:] + bg[h]))
__global__ __launch_bounds__(256) void gate_kernel(const float* __restrict__ x,
                                                   const float* __restrict__ Wg,
                                                   const float* __restrict__ bg,
                                                   float* __restrict__ g) {
  const int row = blockIdx.x;
  const int tid = threadIdx.x;
  const int h = tid >> 4;
  const int seg = tid & 15;
  const float* xr = x + (size_t)row * DIMC;
  const float* wr = Wg + (size_t)h * DIMC;
  float p = 0.f;
  for (int k = seg * 64; k < seg * 64 + 64; ++k) p += xr[k] * wr[k];
  __shared__ float red[256];
  red[tid] = p;
  __syncthreads();
  if (tid < NH) {
    float s = bg[tid];
    for (int i = 0; i < 16; ++i) s += red[tid * 16 + i];
    float sp = (s > 20.f) ? s : log1pf(__expf(s));
    g[(size_t)row * NH + tid] = __expf(-sp);
  }
}

// ---------------------------------------------------------------- scan
// One block per (b,h); lane e owns state column M[:,e] (64 fp32 regs).
// M_t = g_t*M_{t-1} + k_t v_t^T ; h_t[e] = sum_d q_t[d]*M_t[d][e]
__global__ __launch_bounds__(64) void scan_kernel(const float* __restrict__ sq,
                                                  const float* __restrict__ sk,
                                                  const float* __restrict__ sv,
                                                  const float* __restrict__ g,
                                                  float* __restrict__ Hs) {
  const int bh = blockIdx.x;
  const int b = bh / NH, h = bh % NH;
  const int e = threadIdx.x;
  const int colbase = h * HD;
  __shared__ float lk[64], lq[64];
  float m[64];
#pragma unroll
  for (int d = 0; d < 64; ++d) m[d] = 0.f;

  for (int t = 0; t < TSEQ; ++t) {
    const size_t row = (size_t)(b * TSEQ + t) * DIMC;
    lk[e] = sk[row + colbase + e];
    lq[e] = sq[row + colbase + e];
    __syncthreads();
    const float ve = sv[row + colbase + e];
    const float gt = g[(size_t)(b * TSEQ + t) * NH + h];
    float acc = 0.f;
#pragma unroll
    for (int d = 0; d < 64; ++d) {
      m[d] = gt * m[d] + lk[d] * ve;
      acc += lq[d] * m[d];
    }
    Hs[row + colbase + e] = acc;
    __syncthreads();
  }
}

// ---------------------------------------------------------------- l2norm
__global__ __launch_bounds__(64) void l2norm_f16_kernel(const float* __restrict__ X,
                                                        _Float16* __restrict__ Yn) {
  const int row = blockIdx.x;
  const int h = blockIdx.y;
  const int d = threadIdx.x;
  const size_t idx = (size_t)row * DIMC + h * HD + d;
  const float v = X[idx];
  __shared__ float red[64];
  red[d] = v * v;
  __syncthreads();
  for (int s = 32; s > 0; s >>= 1) {
    if (d < s) red[d] += red[d + s];
    __syncthreads();
  }
  float n = fmaxf(sqrtf(red[0]), 1e-12f);
  Yn[idx] = (_Float16)(v / n);
}

// ---------------------------------------------------------------- attention
// Block = (bh, qtile); 128 thr = 4 waves, each wave: 16 q rows x full D=64.
// Flash-style online softmax; QK^T and P@V both on v_wmma_f32_16x16x32_f16.
__global__ __launch_bounds__(128) void attn_kernel(const _Float16* __restrict__ Qn,
                                                   const _Float16* __restrict__ Kn,
                                                   const _Float16* __restrict__ Vb,
                                                   const float* __restrict__ temp,
                                                   _Float16* __restrict__ Out) {
  __shared__ alignas(16) _Float16 lQ[64][64];
  __shared__ alignas(16) _Float16 lK[64][64];
  __shared__ alignas(16) _Float16 lV[64][64];      // transposed: [d][key]
  __shared__ alignas(16) _Float16 lP[4][16][64];   // per-wave P staging

  const int tid = threadIdx.x;
  const int lane = tid & 31;
  const int w = tid >> 5;
  const int bh = blockIdx.x;
  const int qt = blockIdx.y;
  const int b = bh >> 4, h = bh & 15;

  const float tv = temp[h];
  const float scale = (tv > 20.f) ? tv : log1pf(__expf(tv));  // softplus(temp)

  {  // stage Q tile once (batched: loads in flight, then stores)
    int row = tid >> 1;
    int seg = (tid & 1) * 32;
    const v8h* gq = (const v8h*)(Qn + (size_t)(b * TSEQ + qt * 64 + row) * DIMC +
                                 h * HD + seg);
    v8h q0 = gq[0], q1 = gq[1], q2 = gq[2], q3 = gq[3];
    v8h* dq = (v8h*)(&lQ[row][seg]);
    dq[0] = q0; dq[1] = q1; dq[2] = q2; dq[3] = q3;
  }
  __syncthreads();

  const int am = lane & 15;
  const int ak = (lane < 16) ? 0 : 8;
  const int bn = lane & 15;
  const int bk = (lane < 16) ? 0 : 8;
  const int chalf = lane >> 4;

  v16h aQ[2];
#pragma unroll
  for (int st = 0; st < 2; ++st) {
    v8h lo = *(const v8h*)(&lQ[w * 16 + am][st * 32 + ak]);
    v8h hi = *(const v8h*)(&lQ[w * 16 + am][st * 32 + ak + 16]);
    aQ[st] = pack16(lo, hi);
  }

  float mrow[8], lrow[8];
#pragma unroll
  for (int r = 0; r < 8; ++r) { mrow[r] = -3.0e38f; lrow[r] = 0.f; }
  const v8f zero = {0.f, 0.f, 0.f, 0.f, 0.f, 0.f, 0.f, 0.f};
  v8f o[4];
#pragma unroll
  for (int d = 0; d < 4; ++d) o[d] = zero;

  for (int kc = 0; kc < TSEQ / 64; ++kc) {
    {  // stage K chunk (async when available) and transposed V chunk
      int key = tid >> 1;
      int seg = (tid & 1) * 32;
      const size_t grow = (size_t)(b * TSEQ + kc * 64 + key) * DIMC + h * HD + seg;
#if USE_ASYNC_LDS
      const _Float16* gk = Kn + grow;
      async_copy16(gk,      &lK[key][seg]);
      async_copy16(gk + 8,  &lK[key][seg + 8]);
      async_copy16(gk + 16, &lK[key][seg + 16]);
      async_copy16(gk + 24, &lK[key][seg + 24]);
#else
      const v8h* gk = (const v8h*)(Kn + grow);
      v8h k0 = gk[0], k1 = gk[1], k2 = gk[2], k3 = gk[3];
#endif
      const v8h* gv = (const v8h*)(Vb + grow);
      v8h v0 = gv[0], v1 = gv[1], v2 = gv[2], v3 = gv[3];
#if !USE_ASYNC_LDS
      v8h* dk = (v8h*)(&lK[key][seg]);
      dk[0] = k0; dk[1] = k1; dk[2] = k2; dk[3] = k3;
#endif
#pragma unroll
      for (int i = 0; i < 8; ++i) {
        lV[seg + i][key] = v0[i];
        lV[seg + 8 + i][key] = v1[i];
        lV[seg + 16 + i][key] = v2[i];
        lV[seg + 24 + i][key] = v3[i];
      }
#if USE_ASYNC_LDS
      async_wait0();
#endif
    }
    __syncthreads();

    // scores: 16 q rows x 64 keys (4 N-subtiles, K=64 over d in 2 steps)
    v8f s[4];
#pragma unroll
    for (int ns = 0; ns < 4; ++ns) {
      v8f c = zero;
#pragma unroll
      for (int st = 0; st < 2; ++st) {
        v8h lo = *(const v8h*)(&lK[ns * 16 + bn][st * 32 + bk]);
        v8h hi = *(const v8h*)(&lK[ns * 16 + bn][st * 32 + bk + 16]);
        c = wmma_f16(aQ[st], pack16(lo, hi), c);
      }
      s[ns] = c;
    }

    // online softmax per row (row M = r + 8*chalf lives on 16-lane half-group)
#pragma unroll
    for (int r = 0; r < 8; ++r) {
#pragma unroll
      for (int ns = 0; ns < 4; ++ns) s[ns][r] *= scale;
      float mx = fmaxf(fmaxf(s[0][r], s[1][r]), fmaxf(s[2][r], s[3][r]));
#pragma unroll
      for (int off = 1; off < 16; off <<= 1)
        mx = fmaxf(mx, __shfl_xor(mx, off, 32));
      float mnew = fmaxf(mrow[r], mx);
      float corr = __expf(mrow[r] - mnew);
      mrow[r] = mnew;
      lrow[r] *= corr;
#pragma unroll
      for (int d = 0; d < 4; ++d) o[d][r] *= corr;
      float rs = 0.f;
#pragma unroll
      for (int ns = 0; ns < 4; ++ns) {
        float p = __expf(s[ns][r] - mnew);
        s[ns][r] = p;
        rs += p;
      }
#pragma unroll
      for (int off = 1; off < 16; off <<= 1)
        rs += __shfl_xor(rs, off, 32);
      lrow[r] += rs;
    }

    // C-layout -> A-layout bounce through this wave's private LDS region
#pragma unroll
    for (int ns = 0; ns < 4; ++ns)
#pragma unroll
      for (int r = 0; r < 8; ++r)
        lP[w][r + chalf * 8][ns * 16 + (lane & 15)] = (_Float16)s[ns][r];

    // out += P @ V   (contract over keys; B from transposed V tile)
#pragma unroll
    for (int st = 0; st < 2; ++st) {
      v8h plo = *(const v8h*)(&lP[w][am][st * 32 + ak]);
      v8h phi = *(const v8h*)(&lP[w][am][st * 32 + ak + 16]);
      v16h pa = pack16(plo, phi);
#pragma unroll
      for (int ds = 0; ds < 4; ++ds) {
        v8h lo = *(const v8h*)(&lV[ds * 16 + bn][st * 32 + bk]);
        v8h hi = *(const v8h*)(&lV[ds * 16 + bn][st * 32 + bk + 16]);
        o[ds] = wmma_f16(pa, pack16(lo, hi), o[ds]);
      }
    }
    __syncthreads();
  }

#pragma unroll
  for (int ds = 0; ds < 4; ++ds)
#pragma unroll
    for (int r = 0; r < 8; ++r) {
      int row = b * TSEQ + qt * 64 + w * 16 + r + chalf * 8;
      int col = h * HD + ds * 16 + (lane & 15);
      Out[(size_t)row * DIMC + col] = (_Float16)(o[ds][r] / lrow[r]);
    }
}

// ---------------------------------------------------------------- launch
extern "C" void kernel_launch(void* const* d_in, const int* in_sizes, int n_in,
                              void* d_out, int out_size, void* d_ws, size_t ws_size,
                              hipStream_t stream) {
  (void)in_sizes; (void)n_in; (void)out_size; (void)ws_size;

  const float* x    = (const float*)d_in[0];
  const float* Wsq  = (const float*)d_in[1];
  const float* bsq  = (const float*)d_in[2];
  const float* Wsk  = (const float*)d_in[3];
  const float* bsk  = (const float*)d_in[4];
  const float* Wsv  = (const float*)d_in[5];
  const float* bsv  = (const float*)d_in[6];
  const float* Wg   = (const float*)d_in[7];
  const float* bg   = (const float*)d_in[8];
  const float* Wq   = (const float*)d_in[9];
  const float* bq   = (const float*)d_in[10];
  const float* Wk   = (const float*)d_in[11];
  const float* bk_  = (const float*)d_in[12];
  const float* Wv   = (const float*)d_in[13];
  const float* bv   = (const float*)d_in[14];
  const float* Wo   = (const float*)d_in[15];
  const float* bo   = (const float*)d_in[16];
  const float* temp = (const float*)d_in[17];

  char* ws = (char*)d_ws;
  size_t off = 0;
  auto alloc = [&](size_t bytes) {
    void* p = ws + off;
    off += (bytes + 255) & ~(size_t)255;
    return p;
  };

  _Float16* xb   = (_Float16*)alloc((size_t)NT * DIMC * 2);
  _Float16* Wsqb = (_Float16*)alloc((size_t)DIMC * DIMC * 2);
  _Float16* Wskb = (_Float16*)alloc((size_t)DIMC * DIMC * 2);
  _Float16* Wsvb = (_Float16*)alloc((size_t)DIMC * DIMC * 2);
  _Float16* Wqb  = (_Float16*)alloc((size_t)DIMC * DIMC * 2);
  _Float16* Wkb  = (_Float16*)alloc((size_t)DIMC * DIMC * 2);
  _Float16* Wvb  = (_Float16*)alloc((size_t)DIMC * DIMC * 2);
  _Float16* Wob  = (_Float16*)alloc((size_t)DIMC * DIMC * 2);
  float* sq   = (float*)alloc((size_t)NT * DIMC * 4);
  float* sk   = (float*)alloc((size_t)NT * DIMC * 4);
  float* sv   = (float*)alloc((size_t)NT * DIMC * 4);
  float* gbuf = (float*)alloc((size_t)NT * NH * 4);
  float* Hs   = (float*)alloc((size_t)NT * DIMC * 4);
  _Float16* Hb  = (_Float16*)alloc((size_t)NT * DIMC * 2);
  _Float16* Qnh = (_Float16*)alloc((size_t)NT * DIMC * 2);
  _Float16* Knh = (_Float16*)alloc((size_t)NT * DIMC * 2);
  _Float16* Vh  = (_Float16*)alloc((size_t)NT * DIMC * 2);
  // reuse (stream-ordered, producers done before overwrite):
  float* Qf = sq;         // Q projection overwrites sq after scan
  float* Kf = sk;
  float* Vf = sv;
  _Float16* attn = xb;    // attention output reuses x-f16 (x no longer needed)

  const int NE = NT * DIMC;       // 4M elements
  const int WE = DIMC * DIMC;     // 1M elements
  dim3 ggrid(DIMC / 128, NT / 128);  // (8, 32)

  // 1) f32 -> f16 conversions (activations + the 7 GEMM weights)
  f32_to_f16_kernel<<<1024, 256, 0, stream>>>(x, xb, NE);
  f32_to_f16_kernel<<<512, 256, 0, stream>>>(Wsq, Wsqb, WE);
  f32_to_f16_kernel<<<512, 256, 0, stream>>>(Wsk, Wskb, WE);
  f32_to_f16_kernel<<<512, 256, 0, stream>>>(Wsv, Wsvb, WE);
  f32_to_f16_kernel<<<512, 256, 0, stream>>>(Wq, Wqb, WE);
  f32_to_f16_kernel<<<512, 256, 0, stream>>>(Wk, Wkb, WE);
  f32_to_f16_kernel<<<512, 256, 0, stream>>>(Wv, Wvb, WE);
  f32_to_f16_kernel<<<512, 256, 0, stream>>>(Wo, Wob, WE);

  // 2) pre-scan projections (WMMA GEMMs)
  gemm_xwT<<<ggrid, 256, 0, stream>>>(xb, Wsqb, bsq, sq, NT, DIMC, DIMC);
  gemm_xwT<<<ggrid, 256, 0, stream>>>(xb, Wskb, bsk, sk, NT, DIMC, DIMC);
  gemm_xwT<<<ggrid, 256, 0, stream>>>(xb, Wsvb, bsv, sv, NT, DIMC, DIMC);

  // 3) gate + gated-state scan (fp32, 64 parallel (b,h) recurrences)
  gate_kernel<<<NT, 256, 0, stream>>>(x, Wg, bg, gbuf);
  scan_kernel<<<BB * NH, 64, 0, stream>>>(sq, sk, sv, gbuf, Hs);
  f32_to_f16_kernel<<<1024, 256, 0, stream>>>(Hs, Hb, NE);

  // 4) Q/K from Hseq, V from x (WMMA GEMMs; outputs overwrite sq/sk/sv)
  gemm_xwT<<<ggrid, 256, 0, stream>>>(Hb, Wqb, bq, Qf, NT, DIMC, DIMC);
  gemm_xwT<<<ggrid, 256, 0, stream>>>(Hb, Wkb, bk_, Kf, NT, DIMC, DIMC);
  gemm_xwT<<<ggrid, 256, 0, stream>>>(xb, Wvb, bv, Vf, NT, DIMC, DIMC);

  // 5) per-head L2 norm of Q,K (fp32 -> f16), V -> f16
  l2norm_f16_kernel<<<dim3(NT, NH), 64, 0, stream>>>(Qf, Qnh);
  l2norm_f16_kernel<<<dim3(NT, NH), 64, 0, stream>>>(Kf, Knh);
  f32_to_f16_kernel<<<1024, 256, 0, stream>>>(Vf, Vh, NE);

  // 6) flash-style softmax attention (WMMA QK^T and P@V)
  attn_kernel<<<dim3(BB * NH, TSEQ / 64), 128, 0, stream>>>(Qnh, Knh, Vh, temp, attn);

  // 7) output projection into d_out (fp32)
  gemm_xwT<<<ggrid, 256, 0, stream>>>(attn, Wob, bo, (float*)d_out, NT, DIMC, DIMC);
}
